// LinearAttention_16501264351379
// MI455X (gfx1250) — matrix operand order repaired and checked by reading, not compiled, not run on hardware
//
#include <hip/hip_runtime.h>
#include <hip/hip_bf16.h>

#define DIM   256
#define HEADS 8
#define DH    32
#define O3    768
#define NBATCH 8
#define NPIX  4096
#define NPIX_ALL (NBATCH*NPIX)
#define SCALE_Q 0.17677669529663687f
#define EPS_LN  1e-5f

typedef __attribute__((ext_vector_type(16))) _Float16 v16h;
typedef __attribute__((ext_vector_type(8)))  _Float16 v8h;
typedef __attribute__((ext_vector_type(8)))  float    v8f;

union AFrag { v16h v; v8h h8[2]; };

__device__ __forceinline__ v8f wmma_f16(const AFrag& a, v16h b, v8f c) {
    return __builtin_amdgcn_wmma_f32_16x16x32_f16(false, a.v, false, b,
                                                  (short)0, c, false, false);
}

// ---------------- LayerNorm over channels (pre), emit f16 [pixel][C] ----------------
__global__ void k_ln_pre(const float* __restrict__ x, const float* __restrict__ g,
                         _Float16* __restrict__ xn) {
    int p = blockIdx.x * 256 + threadIdx.x;        // global pixel 0..32767
    int b = p >> 12, n = p & 4095;
    const float* xp = x + (size_t)b * DIM * NPIX + n;
    float s = 0.f, sq = 0.f;
    for (int c = 0; c < DIM; ++c) { float v = xp[(size_t)c * NPIX]; s += v; sq += v * v; }
    float mean = s * (1.0f / DIM);
    float var  = sq * (1.0f / DIM) - mean * mean;
    float rstd = rsqrtf(var + EPS_LN);
    _Float16* o = xn + (size_t)p * DIM;
    for (int c = 0; c < DIM; ++c) {
        float v = xp[(size_t)c * NPIX];
        o[c] = (_Float16)((v - mean) * rstd * g[c]);
    }
}

// ---------------- Convert weights to f16 ----------------
__global__ void k_cvt_w(const float* __restrict__ wq, const float* __restrict__ wo,
                        _Float16* __restrict__ wqh, _Float16* __restrict__ woh) {
    int i = blockIdx.x * 256 + threadIdx.x;        // 0..262143
    if (i < O3 * DIM) wqh[i] = (_Float16)wq[i];
    else              woh[i - O3 * DIM] = (_Float16)wo[i - O3 * DIM];
}

// ------- qkv GEMM: [768x256] x [256x32768] ------------------------------------------
// Block = 128x64 tile: 4 waves x (32 rows each), all waves share one 64-col group
// (identical B fragments -> WGP$ hits). A/B fragments double-buffered across kt.
__global__ void k_gemm_qkv(const _Float16* __restrict__ Wq, const _Float16* __restrict__ xn,
                           float* __restrict__ qkv) {
    int wid  = threadIdx.x >> 5, lane = threadIdx.x & 31;
    int smt  = (blockIdx.x % 6) * 4 + wid;         // 0..23 (32-row supertile)
    int snt  = blockIdx.x / 6;                     // 0..511 (64-col supertile)
    int half = lane >> 4, l16 = lane & 15;
    int kb   = 16 * half;
    const _Float16* arow0 = Wq + (size_t)(smt * 32 + l16) * DIM;
    const _Float16* arow1 = arow0 + (size_t)16 * DIM;
    const _Float16* bcol  = xn + (size_t)(snt * 64 + l16) * DIM + 16 * half;

    AFrag a0, a1;
    v16h bf[4];
    a0.h8[0] = *(const v8h*)(arow0 + kb);
    a0.h8[1] = *(const v8h*)(arow0 + kb + 8);
    a1.h8[0] = *(const v8h*)(arow1 + kb);
    a1.h8[1] = *(const v8h*)(arow1 + kb + 8);
    #pragma unroll
    for (int ni = 0; ni < 4; ++ni)
        bf[ni] = *(const v16h*)(bcol + (size_t)ni * 16 * DIM);

    v8f acc[2][4] = {};
    #pragma unroll
    for (int kt = 0; kt < 8; ++kt) {
        AFrag na0, na1;
        v16h nbf[4];
        if (kt < 7) {                               // issue next-iter loads first
            const _Float16* pa0 = arow0 + (kt + 1) * 32;
            const _Float16* pa1 = arow1 + (kt + 1) * 32;
            na0.h8[0] = *(const v8h*)(pa0 + kb);
            na0.h8[1] = *(const v8h*)(pa0 + kb + 8);
            na1.h8[0] = *(const v8h*)(pa1 + kb);
            na1.h8[1] = *(const v8h*)(pa1 + kb + 8);
            #pragma unroll
            for (int ni = 0; ni < 4; ++ni)
                nbf[ni] = *(const v16h*)(bcol + (size_t)ni * 16 * DIM + (kt + 1) * 32);
        }
        #pragma unroll
        for (int ni = 0; ni < 4; ++ni) {
            acc[0][ni] = wmma_f16(a0, bf[ni], acc[0][ni]);
            acc[1][ni] = wmma_f16(a1, bf[ni], acc[1][ni]);
        }
        if (kt < 7) {
            a0 = na0; a1 = na1;
            #pragma unroll
            for (int ni = 0; ni < 4; ++ni) bf[ni] = nbf[ni];
        }
    }
    int base = snt * 64;                 // 64 | 4096 -> whole group in one batch image
    int b = base >> 12, nbase = base & 4095;
    float* outb = qkv + (size_t)b * O3 * NPIX;
    #pragma unroll
    for (int ni = 0; ni < 4; ++ni) {
        int n = nbase + ni * 16 + l16;
        #pragma unroll
        for (int mi = 0; mi < 2; ++mi) {
            #pragma unroll
            for (int j = 0; j < 8; ++j) {
                int o = smt * 32 + mi * 16 + j + 8 * half;
                outb[(size_t)o * NPIX + n] = acc[mi][ni][j];
            }
        }
    }
}

// ---------------- softmax over d (q), emit f16 [bh][n][32], *SCALE ----------------
__global__ void k_softmax_q(const float* __restrict__ qkv, _Float16* __restrict__ qf) {
    int bh = blockIdx.x >> 4, chunk = blockIdx.x & 15;
    int n  = chunk * 256 + threadIdx.x;
    int b  = bh >> 3, h = bh & 7;
    const float* src = qkv + ((size_t)b * O3 + h * DH) * NPIX + n;
    float v[32];
    float m = -1e30f;
    #pragma unroll
    for (int d = 0; d < 32; ++d) { v[d] = src[(size_t)d * NPIX]; m = fmaxf(m, v[d]); }
    float s = 0.f;
    #pragma unroll
    for (int d = 0; d < 32; ++d) { v[d] = expf(v[d] - m); s += v[d]; }
    float inv = SCALE_Q / s;
    _Float16* o = qf + ((size_t)bh * NPIX + n) * DH;
    #pragma unroll
    for (int d = 0; d < 32; ++d) o[d] = (_Float16)(v[d] * inv);
}

// ---------------- softmax over n (k), emit f16 [b][hd][4096] ----------------
__global__ void k_softmax_k(const float* __restrict__ qkv, _Float16* __restrict__ kf) {
    __shared__ float red[256];
    int row = blockIdx.x;                          // 0..2047 == b*256+hd
    int b = row >> 8, hd = row & 255;
    const float* src = qkv + ((size_t)b * O3 + 256 + hd) * NPIX;
    int tid = threadIdx.x;
    float v[16];
    float m = -1e30f;
    #pragma unroll
    for (int i = 0; i < 16; ++i) { v[i] = src[tid + 256 * i]; m = fmaxf(m, v[i]); }
    red[tid] = m; __syncthreads();
    for (int off = 128; off > 0; off >>= 1) {
        if (tid < off) red[tid] = fmaxf(red[tid], red[tid + off]);
        __syncthreads();
    }
    m = red[0]; __syncthreads();
    float s = 0.f;
    #pragma unroll
    for (int i = 0; i < 16; ++i) { v[i] = expf(v[i] - m); s += v[i]; }
    red[tid] = s; __syncthreads();
    for (int off = 128; off > 0; off >>= 1) {
        if (tid < off) red[tid] += red[tid + off];
        __syncthreads();
    }
    float inv = 1.0f / red[0];
    _Float16* o = kf + (size_t)row * NPIX;
    #pragma unroll
    for (int i = 0; i < 16; ++i) o[tid + 256 * i] = (_Float16)(v[i] * inv);
}

// ---------------- convert v to f16 [b][hd][4096] ----------------
__global__ void k_cvt_v(const float* __restrict__ qkv, _Float16* __restrict__ vf) {
    int i = blockIdx.x * 256 + threadIdx.x;        // 0..8388607
    int b = i >> 20, rem = i & 1048575;            // rem = hd*4096+n
    vf[i] = (_Float16)qkv[((size_t)b * O3 + 512) * NPIX + rem];
}

// ------ context partials: k (32x4096) x v^T, K split 4 ways -> f32 [kc][bh][d][e] ---
__global__ void k_context(const _Float16* __restrict__ kf, const _Float16* __restrict__ vf,
                          float* __restrict__ ctxp) {
    int wid  = threadIdx.x >> 5, lane = threadIdx.x & 31;
    int t    = blockIdx.x * 4 + wid;                // 0..1023
    int tt   = t & 3;                               // tile within (b,h)
    int kc   = (t >> 2) & 3;                        // K chunk (1024 each)
    int bh   = t >> 4;                              // 0..63
    int b    = bh >> 3, h = bh & 7;
    int dt   = tt & 1, et = tt >> 1;
    int half = lane >> 4, l16 = lane & 15;
    int drow = dt * 16 + l16;
    int ecol = et * 16 + l16;
    int kb   = 16 * half;
    const _Float16* ka = kf + ((size_t)b * 256 + h * DH + drow) * NPIX + kc * 1024;
    const _Float16* vb = vf + ((size_t)b * 256 + h * DH + ecol) * NPIX + kc * 1024 + 16 * half;
    v8f acc = {};
    for (int ntile = 0; ntile < 32; ++ntile) {
        int nb0 = ntile * 32;
        AFrag a;
        a.h8[0] = *(const v8h*)(ka + nb0 + kb);
        a.h8[1] = *(const v8h*)(ka + nb0 + kb + 8);
        v16h bfr = *(const v16h*)(vb + nb0);
        acc = wmma_f16(a, bfr, acc);
    }
    float* dst = ctxp + (size_t)kc * 65536 + (size_t)bh * 1024;
    #pragma unroll
    for (int j = 0; j < 8; ++j) {
        int d = dt * 16 + j + 8 * half;
        dst[d * 32 + ecol] = acc[j];
    }
}

// ---------------- reduce K-chunk partials -> ctx^T f16 [bh][e][d] ----------------
__global__ void k_ctx_reduce(const float* __restrict__ ctxp, _Float16* __restrict__ ctxT) {
    int i = blockIdx.x * 256 + threadIdx.x;         // 0..65535
    float s = ctxp[i] + ctxp[i + 65536] + ctxp[i + 131072] + ctxp[i + 196608];
    int bh = i >> 10, rem = i & 1023, d = rem >> 5, e = rem & 31;
    ctxT[((size_t)bh * DH + e) * DH + d] = (_Float16)s;
}

// ---- out = ctx^T (32x32) x q (32x4096); wave does 32 rows x 64 cols, f16 [pix][C] --
__global__ void k_attn(const _Float16* __restrict__ ctxT, const _Float16* __restrict__ qf,
                       _Float16* __restrict__ attn) {
    int wid  = threadIdx.x >> 5, lane = threadIdx.x & 31;
    int t    = blockIdx.x * 4 + wid;               // 0..4095
    int ng   = t & 63;                             // group of 64 cols
    int bh   = t >> 6;                             // 0..63
    int half = lane >> 4, l16 = lane & 15;
    int kb   = 16 * half;
    const _Float16* ap0 = ctxT + ((size_t)bh * DH + l16) * DH;
    const _Float16* ap1 = ctxT + ((size_t)bh * DH + 16 + l16) * DH;
    AFrag a0, a1;
    a0.h8[0] = *(const v8h*)(ap0 + kb);
    a0.h8[1] = *(const v8h*)(ap0 + kb + 8);
    a1.h8[0] = *(const v8h*)(ap1 + kb);
    a1.h8[1] = *(const v8h*)(ap1 + kb + 8);
    int b = bh >> 3, h = bh & 7;
    #pragma unroll
    for (int ni = 0; ni < 4; ++ni) {
        int n = ng * 64 + ni * 16 + l16;
        v16h bfr = *(const v16h*)(qf + ((size_t)bh * NPIX + n) * DH + 16 * half);
        v8f c0 = {}, c1 = {};
        c0 = wmma_f16(a0, bfr, c0);
        c1 = wmma_f16(a1, bfr, c1);
        int p = b * NPIX + n;
        v8h p0, p1;
        #pragma unroll
        for (int j = 0; j < 8; ++j) { p0[j] = (_Float16)c0[j]; p1[j] = (_Float16)c1[j]; }
        _Float16* dst = attn + (size_t)p * DIM + h * DH;
        *(v8h*)(dst + 8 * half)      = p0;
        *(v8h*)(dst + 16 + 8 * half) = p1;
    }
}

// ------- out projection GEMM + bias; block = 128x64 tile, B shared by waves --------
__global__ void k_gemm_out(const _Float16* __restrict__ Wo, const _Float16* __restrict__ attn,
                           const float* __restrict__ bias, float* __restrict__ outp) {
    int wid  = threadIdx.x >> 5, lane = threadIdx.x & 31;
    int smt  = (blockIdx.x & 1) * 4 + wid;         // 0..7 (32-row supertile)
    int snt  = blockIdx.x >> 1;                    // 0..511 (64-col supertile)
    int half = lane >> 4, l16 = lane & 15;
    int kb   = 16 * half;
    const _Float16* arow0 = Wo + (size_t)(smt * 32 + l16) * DIM;
    const _Float16* arow1 = arow0 + (size_t)16 * DIM;
    const _Float16* bcol  = attn + (size_t)(snt * 64 + l16) * DIM + 16 * half;

    AFrag a0, a1;
    v16h bf[4];
    a0.h8[0] = *(const v8h*)(arow0 + kb);
    a0.h8[1] = *(const v8h*)(arow0 + kb + 8);
    a1.h8[0] = *(const v8h*)(arow1 + kb);
    a1.h8[1] = *(const v8h*)(arow1 + kb + 8);
    #pragma unroll
    for (int ni = 0; ni < 4; ++ni)
        bf[ni] = *(const v16h*)(bcol + (size_t)ni * 16 * DIM);

    v8f acc[2][4] = {};
    #pragma unroll
    for (int kt = 0; kt < 8; ++kt) {
        AFrag na0, na1;
        v16h nbf[4];
        if (kt < 7) {
            const _Float16* pa0 = arow0 + (kt + 1) * 32;
            const _Float16* pa1 = arow1 + (kt + 1) * 32;
            na0.h8[0] = *(const v8h*)(pa0 + kb);
            na0.h8[1] = *(const v8h*)(pa0 + kb + 8);
            na1.h8[0] = *(const v8h*)(pa1 + kb);
            na1.h8[1] = *(const v8h*)(pa1 + kb + 8);
            #pragma unroll
            for (int ni = 0; ni < 4; ++ni)
                nbf[ni] = *(const v16h*)(bcol + (size_t)ni * 16 * DIM + (kt + 1) * 32);
        }
        #pragma unroll
        for (int ni = 0; ni < 4; ++ni) {
            acc[0][ni] = wmma_f16(a0, bf[ni], acc[0][ni]);
            acc[1][ni] = wmma_f16(a1, bf[ni], acc[1][ni]);
        }
        if (kt < 7) {
            a0 = na0; a1 = na1;
            #pragma unroll
            for (int ni = 0; ni < 4; ++ni) bf[ni] = nbf[ni];
        }
    }
    int base = snt * 64;
    int b = base >> 12, nbase = base & 4095;
    float* outb = outp + (size_t)b * DIM * NPIX;
    #pragma unroll
    for (int ni = 0; ni < 4; ++ni) {
        int n = nbase + ni * 16 + l16;
        #pragma unroll
        for (int mi = 0; mi < 2; ++mi) {
            #pragma unroll
            for (int j = 0; j < 8; ++j) {
                int o = smt * 32 + mi * 16 + j + 8 * half;
                outb[(size_t)o * NPIX + n] = acc[mi][ni][j] + bias[o];
            }
        }
    }
}

// ---------------- final LayerNorm over channels (g only, no bias) ----------------
__global__ void k_ln_post(const float* __restrict__ op, const float* __restrict__ g,
                          float* __restrict__ out) {
    int p = blockIdx.x * 256 + threadIdx.x;
    int b = p >> 12, n = p & 4095;
    const float* xp = op + (size_t)b * DIM * NPIX + n;
    float s = 0.f, sq = 0.f;
    for (int c = 0; c < DIM; ++c) { float v = xp[(size_t)c * NPIX]; s += v; sq += v * v; }
    float mean = s * (1.0f / DIM);
    float var  = sq * (1.0f / DIM) - mean * mean;
    float rstd = rsqrtf(var + EPS_LN);
    float* o = out + (size_t)b * DIM * NPIX + n;
    for (int c = 0; c < DIM; ++c) {
        float v = xp[(size_t)c * NPIX];
        o[(size_t)c * NPIX] = (v - mean) * rstd * g[c];
    }
}

extern "C" void kernel_launch(void* const* d_in, const int* in_sizes, int n_in,
                              void* d_out, int out_size, void* d_ws, size_t ws_size,
                              hipStream_t stream) {
    (void)in_sizes; (void)n_in; (void)out_size; (void)ws_size;
    const float* x     = (const float*)d_in[0];
    const float* g_pre = (const float*)d_in[1];
    const float* w_qkv = (const float*)d_in[2];
    const float* w_out = (const float*)d_in[3];
    const float* b_out = (const float*)d_in[4];
    const float* g_out = (const float*)d_in[5];
    float* out = (float*)d_out;

    char* ws = (char*)d_ws;
    size_t off = 0;
    auto carve = [&](size_t bytes) -> char* {
        char* p = ws + off;
        off += (bytes + 255) & ~(size_t)255;
        return p;
    };
    _Float16* xn   = (_Float16*)carve((size_t)NPIX_ALL * DIM * 2);      // 16.8 MB
    float*    qkv  = (float*)   carve((size_t)NBATCH * O3 * NPIX * 4);  // 100.7 MB
    _Float16* qf   = (_Float16*)carve((size_t)NBATCH * HEADS * NPIX * DH * 2);
    _Float16* kf   = (_Float16*)carve((size_t)NBATCH * DIM * NPIX * 2);
    _Float16* vf   = (_Float16*)carve((size_t)NBATCH * DIM * NPIX * 2);
    float*    ctxp = (float*)   carve((size_t)4 * 64 * DH * DH * 4);    // 1 MB
    _Float16* ctxT = (_Float16*)carve((size_t)NBATCH * HEADS * DH * DH * 2);
    _Float16* attn = (_Float16*)carve((size_t)NPIX_ALL * DIM * 2);
    float*    op   = (float*)   carve((size_t)NBATCH * DIM * NPIX * 4);
    _Float16* wqh  = (_Float16*)carve((size_t)O3 * DIM * 2);
    _Float16* woh  = (_Float16*)carve((size_t)DIM * DIM * 2);

    k_ln_pre    <<<128,  256, 0, stream>>>(x, g_pre, xn);
    k_cvt_w     <<<1024, 256, 0, stream>>>(w_qkv, w_out, wqh, woh);
    k_gemm_qkv  <<<3072, 128, 0, stream>>>(wqh, xn, qkv);
    k_softmax_q <<<1024, 256, 0, stream>>>(qkv, qf);
    k_softmax_k <<<2048, 256, 0, stream>>>(qkv, kf);
    k_cvt_v     <<<32768,256, 0, stream>>>(qkv, vf);
    k_context   <<<256,  128, 0, stream>>>(kf, vf, ctxp);
    k_ctx_reduce<<<256,  256, 0, stream>>>(ctxp, ctxT);
    k_attn      <<<1024, 128, 0, stream>>>(ctxT, qf, attn);
    k_gemm_out  <<<1024, 128, 0, stream>>>(woh, attn, b_out, op);
    k_ln_post   <<<128,  256, 0, stream>>>(op, g_out, out);
}
